// Model_20315195310531
// MI455X (gfx1250) — matrix operand order repaired and checked by reading, not compiled, not run on hardware
//
#include <hip/hip_runtime.h>
#include <hip/hip_bf16.h>

// ---------------------------------------------------------------------------
// 2-layer tanh RNN, B=128, T=1024, H=768.
// Strategy: batch-parallel persistent kernel. 8 workgroups x 16 batch rows,
// each WG runs the full T-loop independently (no inter-WGP sync).
// Weights pre-converted to bf16 in d_ws (L2-resident, 3.4MB), h-state in LDS,
// GEMMs via v_wmma_f32_16x16x32_bf16 with f32 accumulation.
// ---------------------------------------------------------------------------

#define TT    1024
#define H     768
#define KTILES 24          // 768 / 32 (K per WMMA)
#define NTILES 48          // 768 / 16
#define WAVES  16
#define TPB    512         // 16 wave32 waves

typedef __attribute__((ext_vector_type(8)))  __bf16 bf16x8;
typedef __attribute__((ext_vector_type(16))) __bf16 bf16x16;
typedef __attribute__((ext_vector_type(8)))  float  f32x8;

union BV {
    bf16x16 v;
    bf16x8  h[2];
};

__device__ __forceinline__ float fast_tanh(float x) {
    // tanh(x) = 1 - 2/(exp(2x)+1); saturates correctly at +/-inf
    float e = __expf(2.0f * x);
    return 1.0f - 2.0f / (e + 1.0f);
}

// ---------------------------------------------------------------------------
// Prep: convert Whh1, Wih2, Whh2 to bf16; build zero-padded 16x768 Wlin.
// Workspace layout (bf16 elements):
//   [0, M)      Whh1     M = 768*768
//   [M, 2M)     Wih2
//   [2M, 3M)    Whh2
//   [3M, 3M+16*768) WlinPad (rows 0,1 = W_lin; rows 2..15 = 0)
// ---------------------------------------------------------------------------
__global__ void prep_kernel(const float* __restrict__ Whh1,
                            const float* __restrict__ Wih2,
                            const float* __restrict__ Whh2,
                            const float* __restrict__ Wlin,
                            unsigned short* __restrict__ wsu) {
    __bf16* out = (__bf16*)wsu;
    long i = (long)blockIdx.x * 256 + threadIdx.x;
    const long M = 768L * 768L;
    if (i < M) {
        out[i] = (__bf16)Whh1[i];
    } else if (i < 2 * M) {
        out[i] = (__bf16)Wih2[i - M];
    } else if (i < 3 * M) {
        out[i] = (__bf16)Whh2[i - 2 * M];
    } else if (i < 3 * M + 16L * H) {
        long j = i - 3 * M;
        int row = (int)(j / H), k = (int)(j % H);
        out[i] = (row < 2) ? (__bf16)Wlin[row * H + k] : (__bf16)0.0f;
    }
}

// ---------------------------------------------------------------------------
// Persistent RNN kernel: one WG handles batch rows [blockIdx.x*16, +16).
// ---------------------------------------------------------------------------
__launch_bounds__(TPB, 1)
__global__ void rnn_persistent(const float* __restrict__ x,
                               const float* __restrict__ Wih1,
                               const float* __restrict__ bih1,
                               const float* __restrict__ bhh1,
                               const float* __restrict__ bih2,
                               const float* __restrict__ bhh2,
                               const float* __restrict__ blin,
                               const unsigned short* __restrict__ wsu,
                               float* __restrict__ out) {
    const __bf16* Whh1b  = (const __bf16*)wsu;
    const __bf16* Wih2b  = Whh1b + 768L * 768L;
    const __bf16* Whh2b  = Whh1b + 2L * 768L * 768L;
    const __bf16* WlinPb = Whh1b + 3L * 768L * 768L;

    __shared__ __align__(16) __bf16 h1s[16 * H];   // 24 KB
    __shared__ __align__(16) __bf16 h2s[16 * H];   // 24 KB

    const int tid  = threadIdx.x;
    const int lane = tid & 31;
    const int wave = tid >> 5;
    const int half = lane >> 4;          // 0: lanes 0-15, 1: lanes 16-31
    const int l16  = lane & 15;
    const int moff = half * 8;           // C/D row offset for this lane-half
    const int wgb  = blockIdx.x * 16;    // batch base

    // zero-init hidden state
    for (int i = tid; i < 16 * H; i += TPB) {
        h1s[i] = (__bf16)0.0f;
        h2s[i] = (__bf16)0.0f;
    }

    // Per-tile hoisted constants (each wave owns 3 N-tiles).
    int   nt[3];
    float bs1[3], bs2[3], wi0[3], wi1[3];
#pragma unroll
    for (int q = 0; q < 3; ++q) {
        nt[q] = wave + q * WAVES;
        int n = nt[q] * 16 + l16;
        bs1[q] = bih1[n] + bhh1[n];
        bs2[q] = bih2[n] + bhh2[n];
        wi0[q] = Wih1[n * 2 + 0];
        wi1[q] = Wih1[n * 2 + 1];
    }
    const float bl = blin[(l16 < 2) ? l16 : 0];

    __syncthreads();

    // GEMM: acc[q] += A(Asrc 16xH in LDS) x B(Bmat H,H row-major as [N,K])^T
    auto gemm3 = [&](const __bf16* __restrict__ Bmat,
                     const __bf16* __restrict__ Asrc, f32x8* acc) {
#pragma unroll 4
        for (int kt = 0; kt < KTILES; ++kt) {
            // A 16x32 bf16: lane-half K groups {0..7,16..23} / {8..15,24..31}
            BV a;
            const __bf16* ap = Asrc + l16 * H + kt * 32 + half * 8;
            a.h[0] = *(const bf16x8*)(ap);
            a.h[1] = *(const bf16x8*)(ap + 16);
#pragma unroll
            for (int q = 0; q < 3; ++q) {
                // B 32x16: lane-half K groups {0..7,16..23} / {8..15,24..31}
                BV b;
                const __bf16* bp = Bmat + (long)(nt[q] * 16 + l16) * H + kt * 32 + half * 8;
                b.h[0] = *(const bf16x8*)(bp);
                b.h[1] = *(const bf16x8*)(bp + 16);
                acc[q] = __builtin_amdgcn_wmma_f32_16x16x32_bf16(
                    false, a.v, false, b.v, (short)0, acc[q], false, false);
            }
        }
    };

    for (int t = 0; t < TT; ++t) {
        // x inputs for my 8 rows (shared across the 3 tiles)
        float xr0[8], xr1[8];
#pragma unroll
        for (int j = 0; j < 8; ++j) {
            int b = wgb + moff + j;
            xr0[j] = x[(b * 2 + 0) * TT + t];
            xr1[j] = x[(b * 2 + 1) * TT + t];
        }

        // ---------------- layer 1: h1 = tanh(x@Wih1^T + h1@Whh1^T + b) -----
        f32x8 acc[3];
#pragma unroll
        for (int q = 0; q < 3; ++q)
#pragma unroll
            for (int j = 0; j < 8; ++j) acc[q][j] = 0.0f;

        gemm3(Whh1b, h1s, acc);

        __syncthreads();   // all reads of old h1 complete

#pragma unroll
        for (int q = 0; q < 3; ++q) {
            int n = nt[q] * 16 + l16;
#pragma unroll
            for (int j = 0; j < 8; ++j) {
                float v = acc[q][j] + bs1[q] + xr0[j] * wi0[q] + xr1[j] * wi1[q];
                h1s[(moff + j) * H + n] = (__bf16)fast_tanh(v);
            }
        }
        __syncthreads();   // new h1 visible

        // ---------------- layer 2: h2 = tanh(h1@Wih2^T + h2@Whh2^T + b) ----
#pragma unroll
        for (int q = 0; q < 3; ++q)
#pragma unroll
            for (int j = 0; j < 8; ++j) acc[q][j] = 0.0f;

        gemm3(Wih2b, h1s, acc);   // A = new h1
        gemm3(Whh2b, h2s, acc);   // A = old h2

        __syncthreads();   // all reads of old h2 complete

#pragma unroll
        for (int q = 0; q < 3; ++q) {
            int n = nt[q] * 16 + l16;
#pragma unroll
            for (int j = 0; j < 8; ++j) {
                float v = acc[q][j] + bs2[q];
                h2s[(moff + j) * H + n] = (__bf16)fast_tanh(v);
            }
        }
        __syncthreads();   // new h2 visible

        // ---------------- output: out[:, :, t] = h2 @ Wlin^T + b_lin -------
        if (wave == 0) {
            f32x8 oc;
#pragma unroll
            for (int j = 0; j < 8; ++j) oc[j] = 0.0f;
#pragma unroll 4
            for (int kt = 0; kt < KTILES; ++kt) {
                BV a;
                const __bf16* ap = h2s + l16 * H + kt * 32 + half * 8;
                a.h[0] = *(const bf16x8*)(ap);
                a.h[1] = *(const bf16x8*)(ap + 16);
                BV b;
                const __bf16* bp = WlinPb + (long)l16 * H + kt * 32 + half * 8;
                b.h[0] = *(const bf16x8*)(bp);
                b.h[1] = *(const bf16x8*)(bp + 16);
                oc = __builtin_amdgcn_wmma_f32_16x16x32_bf16(
                    false, a.v, false, b.v, (short)0, oc, false, false);
            }
            if (l16 < 2) {   // only output columns 0,1 are real
#pragma unroll
                for (int j = 0; j < 8; ++j) {
                    int b = wgb + moff + j;
                    out[((long)(b * 2 + l16)) * TT + t] = oc[j] + bl;
                }
            }
        }
        // No extra barrier: next step's writes to h1s/h2s are gated by its
        // own barriers, which all occur after wave 0's reads above.
    }
}

// ---------------------------------------------------------------------------
extern "C" void kernel_launch(void* const* d_in, const int* in_sizes, int n_in,
                              void* d_out, int out_size, void* d_ws, size_t ws_size,
                              hipStream_t stream) {
    const float* x    = (const float*)d_in[0];
    const float* Wih1 = (const float*)d_in[1];
    const float* Whh1 = (const float*)d_in[2];
    const float* bih1 = (const float*)d_in[3];
    const float* bhh1 = (const float*)d_in[4];
    const float* Wih2 = (const float*)d_in[5];
    const float* Whh2 = (const float*)d_in[6];
    const float* bih2 = (const float*)d_in[7];
    const float* bhh2 = (const float*)d_in[8];
    const float* Wlin = (const float*)d_in[9];
    const float* blin = (const float*)d_in[10];
    // d_in[11] = future (scalar, == 0 in this config) -> ignored

    unsigned short* wsu = (unsigned short*)d_ws;
    const long M = 768L * 768L;
    long total = 3 * M + 16L * H;                 // 1,781,760 bf16 elements
    int blocks = (int)((total + 255) / 256);
    prep_kernel<<<blocks, 256, 0, stream>>>(Whh1, Wih2, Whh2, Wlin, wsu);

    rnn_persistent<<<128 / 16, TPB, 0, stream>>>(
        x, Wih1, bih1, bhh1, bih2, bhh2, blin, wsu, (float*)d_out);
}
